// FeatureVariableSelection_69587060129839
// MI455X (gfx1250) — matrix operand order repaired and checked by reading, not compiled, not run on hardware
//
#include <hip/hip_runtime.h>

// ---------------- problem constants ----------------
#define NT   16384          // B*S tokens
#define HH   128
#define FF   12
#define KIN  1536           // H*F

typedef unsigned short u16;
typedef __attribute__((ext_vector_type(16))) __bf16 v16bf;
typedef __attribute__((ext_vector_type(8)))  __bf16 v8bf;
typedef __attribute__((ext_vector_type(8)))  float  v8f;

// ---------------- helpers ----------------
__device__ __forceinline__ u16 f2u(float f) {          // f32 -> bf16 (RNE)
    unsigned x = __float_as_uint(f);
    return (u16)((x + 0x7fffu + ((x >> 16) & 1u)) >> 16);
}
__device__ __forceinline__ float u2f(u16 u) { return __uint_as_float(((unsigned)u) << 16); }
__device__ __forceinline__ v8f zero8() {
    v8f z = {0.f,0.f,0.f,0.f,0.f,0.f,0.f,0.f}; return z;
}
__device__ __forceinline__ v8f wmma_bf16(v16bf a, v16bf b, v8f c) {
    return __builtin_amdgcn_wmma_f32_16x16x32_bf16(false, a, false, b, (short)0, c, false, false);
}
// A fragment (16x32) from row-major bf16: two contiguous 16B vector loads per lane
__device__ __forceinline__ v16bf load_a_bf16(const u16* __restrict__ A, long row0,
                                             int lda, int k0, int lane) {
    int nlo = lane & 15, hi8 = (lane >> 4) * 16;
    const u16* p = A + (row0 + nlo) * (long)lda + k0 + hi8;
    v8bf lo = *reinterpret_cast<const v8bf*>(p);
    v8bf hh = *reinterpret_cast<const v8bf*>(p + 8);
    return __builtin_shufflevector(lo, hh, 0,1,2,3,4,5,6,7,8,9,10,11,12,13,14,15);
}

// ---------------- conversion kernels ----------------
// feat (N,H,F) f32 -> flatbf (N,1536) bf16  AND  xbf (F,N,H) bf16 (transposed)
__global__ void k_cvt_feat(const float* __restrict__ feat, u16* __restrict__ flatbf,
                           u16* __restrict__ xbf, long total) {
    long i = (long)blockIdx.x * blockDim.x + threadIdx.x;
    if (i >= total) return;
    u16 v = f2u(feat[i]);
    flatbf[i] = v;
    long n = i / KIN;
    int  r = (int)(i % KIN);
    int  h = r / FF;
    int  f = r % FF;
    xbf[((long)f * NT + n) * HH + h] = v;
}

// weights (batches,K,128) f32 -> swizzled bf16 [b][kt][nt][lane][16] (WMMA B-frag order)
__global__ void k_swz(const float* __restrict__ src, u16* __restrict__ dst,
                      int batches, int K) {
    long i = (long)blockIdx.x * blockDim.x + threadIdx.x;
    long per = (long)K * 128;
    if (i >= (long)batches * per) return;
    int  b = (int)(i / per);
    long r = i % per;
    int e    = (int)(r & 15);
    int lane = (int)((r >> 4) & 31);
    int nt   = (int)((r >> 9) & 7);
    int kt   = (int)(r >> 12);
    int k = kt * 32 + (lane >> 4) * 16 + e;
    int n = nt * 16 + (lane & 15);
    dst[i] = f2u(src[((long)b * K + k) * 128 + n]);
}

// narrow weights (K,cols<=16) f32 -> swizzled bf16 [kt][lane][16], zero-padded to 16 cols
__global__ void k_swz_pad(const float* __restrict__ src, u16* __restrict__ dst,
                          int K, int cols) {
    long i = (long)blockIdx.x * blockDim.x + threadIdx.x;
    long total = (long)(K / 32) * 512;
    if (i >= total) return;
    int e    = (int)(i & 15);
    int lane = (int)((i >> 4) & 31);
    int kt   = (int)(i >> 9);
    int k = kt * 32 + (lane >> 4) * 16 + e;
    int n = lane & 15;
    dst[i] = (n < cols) ? f2u(src[(long)k * cols + n]) : (u16)0;
}

// ---------------- LDS-free WMMA GEMM: C(M,128) = act(A(M,K)*B(K,128)+bias) ----------------
// block: 128 threads = 4 waves; wave w -> rows [m0+16w, +16), all 128 cols (8 wmma tiles)
// B is pre-swizzled [kt][nt][lane][16].  ACT: 0=none, 1=ELU, 2=sigmoid(acc1+b1)*(acc2+b2)
template<int ACT>
__global__ __launch_bounds__(128) void gemm128(
    const u16*  __restrict__ Abase, long aBatch, int lda,
    const u16*  __restrict__ B1,    const u16* __restrict__ B2, long bBatch,
    const float* __restrict__ bias1, const float* __restrict__ bias2, int biasBatch,
    u16* __restrict__ Obase, long oBatch, int K)
{
    const int tid  = threadIdx.x;
    const int lane = tid & 31;
    const int w    = tid >> 5;
    const int batch = blockIdx.y;
    const long m0   = (long)blockIdx.x * 64 + w * 16;

    const u16*   Ab  = Abase + (long)batch * aBatch;
    const v16bf* Bv1 = reinterpret_cast<const v16bf*>(B1 + (long)batch * bBatch);
    const v16bf* Bv2 = (ACT == 2) ? reinterpret_cast<const v16bf*>(B2 + (long)batch * bBatch)
                                  : nullptr;
    const float* bb1 = bias1 + (long)batch * biasBatch;
    const float* bb2 = (ACT == 2) ? (bias2 + (long)batch * biasBatch) : nullptr;
    u16*         Ob  = Obase + (long)batch * oBatch;

    v8f acc[8], acc2[8];
#pragma unroll
    for (int t = 0; t < 8; ++t) { acc[t] = zero8(); if (ACT == 2) acc2[t] = zero8(); }

    const int ktn = K >> 5;
    for (int kt = 0; kt < ktn; ++kt) {
        v16bf a = load_a_bf16(Ab, m0, lda, kt * 32, lane);
        const long fb = ((long)kt * 8) * 32 + lane;
#pragma unroll
        for (int nt = 0; nt < 8; ++nt) {
            v16bf b = Bv1[fb + nt * 32];
            acc[nt] = wmma_bf16(a, b, acc[nt]);
            if (ACT == 2) {
                v16bf b2 = Bv2[fb + nt * 32];
                acc2[nt] = wmma_bf16(a, b2, acc2[nt]);
            }
        }
    }

    const int hi  = lane >> 4;
    const int nlo = lane & 15;
#pragma unroll
    for (int t = 0; t < 8; ++t) {
        int col = t * 16 + nlo;
        float b1v = bb1[col];
        float b2v = (ACT == 2) ? bb2[col] : 0.f;
#pragma unroll
        for (int e = 0; e < 8; ++e) {
            long row = m0 + hi * 8 + e;
            float v = acc[t][e] + b1v;
            if (ACT == 1) v = (v > 0.f) ? v : (__expf(v) - 1.f);
            if (ACT == 2) { float v2 = acc2[t][e] + b2v; v = v2 / (1.f + __expf(-v)); }
            Ob[row * 128 + col] = f2u(v);
        }
    }
}

// ---------------- fused selection tail ----------------
// eta = hid(N,128)@Vo(128,16), skip = flatbf(N,1536)@Vs(1536,16) via WMMA (swizzled B),
// then GLU + LN(12) + softmax -> wout(N,12)
__global__ __launch_bounds__(256) void k_select(
    const u16* __restrict__ flatbf, const u16* __restrict__ hid,
    const u16* __restrict__ VoS, const u16* __restrict__ VsS,
    const float* __restrict__ vbo, const float* __restrict__ vbs,
    const float* __restrict__ Vg,  const float* __restrict__ vbg,
    const float* __restrict__ Vv,  const float* __restrict__ vbv,
    const float* __restrict__ vgamma, const float* __restrict__ vbeta,
    float* __restrict__ wout)
{
    __shared__ float etaS[8][16][16];
    __shared__ float skipS[8][16][16];
    const int tid  = threadIdx.x;
    const int lane = tid & 31;
    const int w    = tid >> 5;
    const int hi   = lane >> 4;
    const int nlo  = lane & 15;
    const long tok0 = (long)blockIdx.x * 128 + w * 16;

    const v16bf* VoV = reinterpret_cast<const v16bf*>(VoS);
    const v16bf* VsV = reinterpret_cast<const v16bf*>(VsS);

    v8f accE = zero8();
#pragma unroll
    for (int kt = 0; kt < 4; ++kt) {
        v16bf a = load_a_bf16(hid, tok0, 128, kt * 32, lane);
        accE = wmma_bf16(a, VoV[kt * 32 + lane], accE);
    }
    v8f accS = zero8();
    for (int kt = 0; kt < 48; ++kt) {
        v16bf a = load_a_bf16(flatbf, tok0, KIN, kt * 32, lane);
        accS = wmma_bf16(a, VsV[kt * 32 + lane], accS);
    }
#pragma unroll
    for (int e = 0; e < 8; ++e) {
        etaS [w][hi * 8 + e][nlo] = accE[e];
        skipS[w][hi * 8 + e][nlo] = accS[e];
    }
    __syncthreads();

    if (tid < 128) {
        const int wv = tid >> 4, r = tid & 15;
        const long tok = (long)blockIdx.x * 128 + tid;
        float e[FF], s[FF], z[FF];
#pragma unroll
        for (int j = 0; j < FF; ++j) {
            e[j] = etaS [wv][r][j] + vbo[j];
            s[j] = skipS[wv][r][j] + vbs[j];
        }
#pragma unroll
        for (int j = 0; j < FF; ++j) {
            float g1 = vbg[j], g2 = vbv[j];
#pragma unroll
            for (int i = 0; i < FF; ++i) { g1 += e[i] * Vg[i * FF + j]; g2 += e[i] * Vv[i * FF + j]; }
            z[j] = s[j] + g2 / (1.f + __expf(-g1));
        }
        float m = 0.f;
#pragma unroll
        for (int j = 0; j < FF; ++j) m += z[j];
        m *= (1.f / FF);
        float var = 0.f;
#pragma unroll
        for (int j = 0; j < FF; ++j) { float d = z[j] - m; var += d * d; }
        var *= (1.f / FF);
        float inv = rsqrtf(var + 1e-5f);
        float mx = -1e30f, l[FF];
#pragma unroll
        for (int j = 0; j < FF; ++j) { l[j] = (z[j] - m) * inv * vgamma[j] + vbeta[j]; mx = fmaxf(mx, l[j]); }
        float ssum = 0.f;
#pragma unroll
        for (int j = 0; j < FF; ++j) { l[j] = __expf(l[j] - mx); ssum += l[j]; }
        float rs = 1.f / ssum;
#pragma unroll
        for (int j = 0; j < FF; ++j) wout[tok * FF + j] = l[j] * rs;
    }
}

// ---------------- per-token LN over H + weighted reduce over F ----------------
__global__ __launch_bounds__(128) void k_finalize(
    const float* __restrict__ feat, const u16* __restrict__ gate,
    const float* __restrict__ wsel, const float* __restrict__ gln,
    const float* __restrict__ bln,  float* __restrict__ selected)
{
    __shared__ float s1[128], s2[128], wsh[FF];
    const int  h = threadIdx.x;
    const long n = blockIdx.x;
    if (h < FF) wsh[h] = wsel[n * FF + h];

    float x[FF];                                         // feat[n, h, 0..11] (48B aligned)
    const float* fp = feat + n * KIN + h * FF;
    *reinterpret_cast<float4*>(x + 0) = *reinterpret_cast<const float4*>(fp + 0);
    *reinterpret_cast<float4*>(x + 4) = *reinterpret_cast<const float4*>(fp + 4);
    *reinterpret_cast<float4*>(x + 8) = *reinterpret_cast<const float4*>(fp + 8);

    float acc = 0.f;
    for (int f = 0; f < FF; ++f) {
        float v = x[f] + u2f(gate[((long)f * NT + n) * 128 + h]);
        __syncthreads();
        s1[h] = v; s2[h] = v * v;
        __syncthreads();
        for (int st = 64; st > 0; st >>= 1) {
            if (h < st) { s1[h] += s1[h + st]; s2[h] += s2[h + st]; }
            __syncthreads();
        }
        float m   = s1[0] * (1.f / 128.f);
        float var = s2[0] * (1.f / 128.f) - m * m;
        float t = (v - m) * rsqrtf(var + 1e-5f) * gln[f * 128 + h] + bln[f * 128 + h];
        acc += wsh[f] * t;
    }
    selected[n * 128 + h] = acc;
}

// ---------------- host launcher ----------------
extern "C" void kernel_launch(void* const* d_in, const int* in_sizes, int n_in,
                              void* d_out, int out_size, void* d_ws, size_t ws_size,
                              hipStream_t stream) {
    const float* feat = (const float*)d_in[0];
    const float* Wi   = (const float*)d_in[1];
    const float* bi   = (const float*)d_in[2];
    const float* Wo   = (const float*)d_in[3];
    const float* bo   = (const float*)d_in[4];
    const float* Wg   = (const float*)d_in[5];
    const float* bg   = (const float*)d_in[6];
    const float* Wv   = (const float*)d_in[7];
    const float* bv   = (const float*)d_in[8];
    const float* gln  = (const float*)d_in[9];
    const float* bln  = (const float*)d_in[10];
    const float* Vi   = (const float*)d_in[11];
    const float* vbi  = (const float*)d_in[12];
    const float* Vo   = (const float*)d_in[13];
    const float* vbo  = (const float*)d_in[14];
    const float* Vg   = (const float*)d_in[15];
    const float* vbg  = (const float*)d_in[16];
    const float* Vv   = (const float*)d_in[17];
    const float* vbv  = (const float*)d_in[18];
    const float* Vs   = (const float*)d_in[19];
    const float* vbs  = (const float*)d_in[20];
    const float* vgm  = (const float*)d_in[21];
    const float* vbt  = (const float*)d_in[22];

    char* ws = (char*)d_ws;
    size_t off = 0;
    u16* flatbf = (u16*)(ws + off); off += (size_t)NT * KIN * 2;         // 50.3 MB
    u16* xbf    = (u16*)(ws + off); off += (size_t)FF * NT * HH * 2;     // 50.3 MB
    u16* ViS    = (u16*)(ws + off); off += (size_t)KIN * 128 * 2;
    u16* VoS    = (u16*)(ws + off); off += (size_t)128 * 16 * 2;
    u16* VsS    = (u16*)(ws + off); off += (size_t)KIN * 16 * 2;
    u16* WiS    = (u16*)(ws + off); off += (size_t)FF * 128 * 128 * 2;
    u16* WoS    = (u16*)(ws + off); off += (size_t)FF * 128 * 128 * 2;
    u16* WgS    = (u16*)(ws + off); off += (size_t)FF * 128 * 128 * 2;
    u16* WvS    = (u16*)(ws + off); off += (size_t)FF * 128 * 128 * 2;
    u16* hidB   = (u16*)(ws + off); off += (size_t)NT * 128 * 2;         // 4.2 MB
    u16* bufA   = (u16*)(ws + off); off += (size_t)FF * NT * HH * 2;     // 50.3 MB
    u16* bufB   = (u16*)(ws + off); off += (size_t)FF * NT * HH * 2;     // 50.3 MB

    float* selected = (float*)d_out;                    // (N,128)
    float* wout     = (float*)d_out + (size_t)NT * HH;  // (N,12) == sparse_weights

    // 1) convert activations once: flat bf16 + transposed per-feature bf16
    {
        long total = (long)NT * KIN;
        k_cvt_feat<<<(unsigned)((total + 255) / 256), 256, 0, stream>>>(feat, flatbf, xbf, total);
    }
    // 2) swizzle weights into WMMA B-fragment order (bf16)
    k_swz<<<768, 256, 0, stream>>>(Vi, ViS, 1, KIN);
    k_swz<<<768, 256, 0, stream>>>(Wi, WiS, FF, 128);
    k_swz<<<768, 256, 0, stream>>>(Wo, WoS, FF, 128);
    k_swz<<<768, 256, 0, stream>>>(Wg, WgS, FF, 128);
    k_swz<<<768, 256, 0, stream>>>(Wv, WvS, FF, 128);
    k_swz_pad<<<8,  256, 0, stream>>>(Vo, VoS, 128, FF);
    k_swz_pad<<<96, 256, 0, stream>>>(Vs, VsS, KIN, FF);

    // 3) part 1: hid = elu(flat @ Vi + vbi)
    gemm128<1><<<dim3(NT / 64, 1), 128, 0, stream>>>(
        flatbf, 0L, KIN, ViS, nullptr, 0L, vbi, nullptr, 0, hidB, 0L, KIN);

    // 4) part 1 tail: eta/skip WMMA + GLU + LN + softmax -> sparse weights
    k_select<<<NT / 128, 256, 0, stream>>>(flatbf, hidB, VoS, VsS, vbo, vbs,
                                           Vg, vbg, Vv, vbv, vgm, vbt, wout);

    // 5) part 2 GRN chain, batched over F (grid.y = 12)
    const long AB = (long)NT * HH, BB = 128L * 128, OB = (long)NT * HH;
    gemm128<1><<<dim3(NT / 64, FF), 128, 0, stream>>>(
        xbf, AB, 128, WiS, nullptr, BB, bi, nullptr, 128, bufA, OB, 128);
    gemm128<0><<<dim3(NT / 64, FF), 128, 0, stream>>>(
        bufA, AB, 128, WoS, nullptr, BB, bo, nullptr, 128, bufB, OB, 128);
    gemm128<2><<<dim3(NT / 64, FF), 128, 0, stream>>>(
        bufB, AB, 128, WgS, WvS, BB, bg, bv, 128, bufA, OB, 128);

    // 6) LN over H + weighted feature reduce -> selected
    k_finalize<<<NT, 128, 0, stream>>>(feat, bufA, wout, gln, bln, selected);

    (void)in_sizes; (void)n_in; (void)out_size; (void)ws_size;
}
